// SelfAttentionBlock_69123203662565
// MI455X (gfx1250) — matrix operand order repaired and checked by reading, not compiled, not run on hardware
//
#include <hip/hip_runtime.h>

typedef __attribute__((ext_vector_type(16))) __bf16 v16bf;
typedef __attribute__((ext_vector_type(8)))  float  v8f;

#define NB   8      // batch
#define CIN  512
#define HW   4096
#define HWP  1024

union PK8  { __bf16 h[8];  uint4 u;    };
union PK16 { __bf16 h[16]; uint4 u[2]; };
union F8   { v8f v; float f[8]; };
union AB16 { v16bf v; uint4 u[2]; };

// ------------------------------------------------------------------
// LDS staging into WMMA fragment order (wave32 layouts, cdna5 CDNA5 ISA)
// A 16x32 bf16 frag: lane L holds row M=L&15;  K(e,L)=e+16*(L>>4)
// B 32x16 bf16 frag: lane L holds col N=L&15;  K(e,L)=e+16*(L>>4)
// ------------------------------------------------------------------

// A source row-major fp32 [.. , lda], stage one 64x32 tile (with f32->bf16 cvt)
__device__ __forceinline__ void stageA_f32(const float* A, int lda, int t, __bf16* ldsA) {
  const int half = t & 1, id = t >> 1, tm = id >> 5, L = id & 31;
  const int m = tm * 16 + (L & 15), h = L >> 4;
  const int k0 = 16 * h + half * 8;            // contiguous 8-elem K segment
  const float* p = A + (size_t)m * lda + k0;
  PK8 pk;
#pragma unroll
  for (int i = 0; i < 8; ++i) pk.h[i] = (__bf16)p[i];
  *(uint4*)&ldsA[((tm * 32 + L) << 4) + half * 8] = pk.u;
}

// B source row-major fp32 [K][ldb], stage one 32x128 tile (with f32->bf16 cvt)
__device__ __forceinline__ void stageB_f32(const float* Bm, int ldb, int t, __bf16* ldsB) {
  const int n = t & 127, kh = t >> 7;
  const float* p = Bm + (size_t)(kh * 16) * ldb + n;
  PK16 pk;
#pragma unroll
  for (int i = 0; i < 16; ++i) pk.h[i] = (__bf16)p[(size_t)i * ldb];
  const int L = (n & 15) + (kh << 4), tn = n >> 4;
  uint4* d = (uint4*)&ldsB[(tn * 32 + L) << 4];
  d[0] = pk.u[0]; d[1] = pk.u[1];
}

// B source row-major bf16 [K][ldb] (strided gather -> swizzled LDS write)
__device__ __forceinline__ void stageB_bf16(const __bf16* Bm, int ldb, int t, __bf16* ldsB) {
  const int n = t & 127, kh = t >> 7;
  const __bf16* p = Bm + (size_t)(kh * 16) * ldb + n;
  PK16 pk;
#pragma unroll
  for (int i = 0; i < 16; ++i) pk.h[i] = p[(size_t)i * ldb];
  const int L = (n & 15) + (kh << 4), tn = n >> 4;
  uint4* d = (uint4*)&ldsB[(tn * 32 + L) << 4];
  d[0] = pk.u[0]; d[1] = pk.u[1];
}

// CDNA5 async global->LDS copy of 16 bytes (ASYNCcnt-tracked, no VGPR round trip)
__device__ __forceinline__ void async_b128(__bf16* ldsDst, const __bf16* gsrc) {
  const unsigned off = (unsigned)(uintptr_t)ldsDst;   // flat LDS addr[31:0] = DS offset
  asm volatile("global_load_async_to_lds_b128 %0, %1, off"
               :: "v"(off), "v"(gsrc) : "memory");
}

// per-wave 2x2 tile MMA over one 32-K step
__device__ __forceinline__ void wave_mma(const __bf16* ldsA, const __bf16* ldsB,
                                         int wm, int wn, int L, F8 acc[2][2]) {
  v16bf a[2], b[2];
#pragma unroll
  for (int i = 0; i < 2; ++i) a[i] = *(const v16bf*)&ldsA[((wm * 2 + i) * 32 + L) << 4];
#pragma unroll
  for (int j = 0; j < 2; ++j) b[j] = *(const v16bf*)&ldsB[((wn * 2 + j) * 32 + L) << 4];
#pragma unroll
  for (int i = 0; i < 2; ++i)
#pragma unroll
    for (int j = 0; j < 2; ++j)
      acc[i][j].v = __builtin_amdgcn_wmma_f32_16x16x32_bf16(
          false, a[i], false, b[j], (short)0, acc[i][j].v, false, false);
}

// ------------------------------------------------------------------
// Kernel 1: 1x1-conv projection  out[o,n] = sum_c W[o,c]*X[c,n]  (per batch)
// ------------------------------------------------------------------
template <bool TRANS_OUT>
__global__ __launch_bounds__(256) void proj_kernel(const float* __restrict__ W,
                                                   const float* __restrict__ X,
                                                   __bf16* __restrict__ out,
                                                   int M, int K) {
  __shared__ __bf16 ldsA[4 * 32 * 16];
  __shared__ __bf16 ldsB[8 * 32 * 16];
  const int b = blockIdx.z, nBase = blockIdx.x * 128, mBase = blockIdx.y * 64;
  const float* Xb = X + (size_t)b * K * HW;
  const int t = threadIdx.x, L = t & 31, w = t >> 5, wm = w >> 2, wn = w & 3;
  F8 acc[2][2];
#pragma unroll
  for (int i = 0; i < 2; ++i)
#pragma unroll
    for (int j = 0; j < 2; ++j)
#pragma unroll
      for (int r = 0; r < 8; ++r) acc[i][j].f[r] = 0.f;
  for (int kb = 0; kb < K; kb += 32) {
    stageA_f32(W + (size_t)mBase * K + kb, K, t, ldsA);
    stageB_f32(Xb + (size_t)kb * HW + nBase, HW, t, ldsB);
    __syncthreads();
    wave_mma(ldsA, ldsB, wm, wn, L, acc);
    __syncthreads();
  }
  const int h = L >> 4, lc = L & 15;
#pragma unroll
  for (int i = 0; i < 2; ++i)
#pragma unroll
    for (int j = 0; j < 2; ++j) {
      const int mB = mBase + (wm * 2 + i) * 16;
      const int nC = nBase + (wn * 2 + j) * 16 + lc;
      if (TRANS_OUT) {   // out[n][m], ld = M  (16B packed store per lane)
        PK8 pk;
#pragma unroll
        for (int r = 0; r < 8; ++r) pk.h[r] = (__bf16)acc[i][j].f[r];
        *(uint4*)(out + (size_t)b * HW * M + (size_t)nC * M + mB + 8 * h) = pk.u;
      } else {           // out[m][n], ld = HW
        __bf16* o = out + (size_t)b * M * HW + (size_t)(mB + 8 * h) * HW + nC;
#pragma unroll
        for (int r = 0; r < 8; ++r) o[(size_t)r * HW] = (__bf16)acc[i][j].f[r];
      }
    }
}

// ------------------------------------------------------------------
// Kernel 2: 2x2 max pool on [B][O][64][64] -> [B][O][32][32] (opt. transposed)
// ------------------------------------------------------------------
__global__ __launch_bounds__(256) void pool_kernel(const __bf16* __restrict__ in,
                                                   __bf16* __restrict__ out,
                                                   int O, int transposed) {
  const int idx = blockIdx.x * 256 + threadIdx.x;
  if (idx >= NB * O * HWP) return;
  const int w2 = idx & 31, h2 = (idx >> 5) & 31;
  const int o = (idx >> 10) % O, b = idx / (O * HWP);
  const __bf16* p = in + (size_t)(b * O + o) * HW + (h2 * 2) * 64 + w2 * 2;
  const float m = fmaxf(fmaxf((float)p[0], (float)p[1]),
                        fmaxf((float)p[64], (float)p[65]));
  const int mpos = h2 * 32 + w2;
  if (transposed) out[((size_t)b * HWP + mpos) * O + o] = (__bf16)m;
  else            out[((size_t)(b * O + o)) * HWP + mpos] = (__bf16)m;
}

// ------------------------------------------------------------------
// Kernel 3: fused scores + softmax -> beta[n][m] bf16.
// Block = 16 query rows x all 1024 keys; wave w owns 128-key strip; K = 64.
// ------------------------------------------------------------------
__global__ __launch_bounds__(256) void scores_softmax_kernel(
    const __bf16* __restrict__ thetaT,   // [B][HW][64]
    const __bf16* __restrict__ phiPT,    // [B][HWP][64]
    __bf16* __restrict__ beta) {         // [B][HW][HWP]
  __shared__ float smred[8][16];
  const int b = blockIdx.y, nBase = blockIdx.x * 16;
  const __bf16* th = thetaT + (size_t)b * HW * 64 + (size_t)nBase * 64;
  const __bf16* ph = phiPT + (size_t)b * HWP * 64;
  __bf16* bout = beta + (size_t)b * HW * HWP + (size_t)nBase * HWP;
  const int t = threadIdx.x, L = t & 31, w = t >> 5;
  const int h = L >> 4, lc = L & 15;

  // A fragments: row n = nBase+lc, contiguous K segments in thetaT
  AB16 afr[2];
#pragma unroll
  for (int s = 0; s < 2; ++s) {
    const __bf16* pa = th + lc * 64 + s * 32 + 16 * h;
    afr[s].u[0] = *(const uint4*)pa;
    afr[s].u[1] = *(const uint4*)(pa + 8);
  }
  F8 acc[8];
#pragma unroll
  for (int tt = 0; tt < 8; ++tt) {
#pragma unroll
    for (int r = 0; r < 8; ++r) acc[tt].f[r] = 0.f;
    const __bf16* pb = ph + (size_t)(w * 128 + tt * 16 + lc) * 64 + 16 * h;
#pragma unroll
    for (int s = 0; s < 2; ++s) {
      AB16 bfr;
      bfr.u[0] = *(const uint4*)(pb + 32 * s);
      bfr.u[1] = *(const uint4*)(pb + 32 * s + 8);
      acc[tt].v = __builtin_amdgcn_wmma_f32_16x16x32_bf16(
          false, afr[s].v, false, bfr.v, (short)0, acc[tt].v, false, false);
    }
  }
  // ---- softmax over m (rows r+8h live in lane-half h, cols in lanes of half) ----
  float red[8];
#pragma unroll
  for (int r = 0; r < 8; ++r) {
    float m = acc[0].f[r];
#pragma unroll
    for (int tt = 1; tt < 8; ++tt) m = fmaxf(m, acc[tt].f[r]);
#pragma unroll
    for (int msk = 1; msk < 16; msk <<= 1) m = fmaxf(m, __shfl_xor(m, msk, 16));
    red[r] = m;
  }
  if (lc == 0) {
#pragma unroll
    for (int r = 0; r < 8; ++r) smred[w][r + 8 * h] = red[r];
  }
  __syncthreads();
  float rowmax[8];
#pragma unroll
  for (int r = 0; r < 8; ++r) {
    float m = -3.0e38f;
    for (int w2 = 0; w2 < 8; ++w2) m = fmaxf(m, smred[w2][r + 8 * h]);
    rowmax[r] = m;
  }
  __syncthreads();
#pragma unroll
  for (int r = 0; r < 8; ++r) red[r] = 0.f;
#pragma unroll
  for (int tt = 0; tt < 8; ++tt)
#pragma unroll
    for (int r = 0; r < 8; ++r) {
      const float e = __expf(acc[tt].f[r] - rowmax[r]);
      acc[tt].f[r] = e;
      red[r] += e;
    }
#pragma unroll
  for (int r = 0; r < 8; ++r) {
    float s = red[r];
#pragma unroll
    for (int msk = 1; msk < 16; msk <<= 1) s += __shfl_xor(s, msk, 16);
    red[r] = s;
  }
  if (lc == 0) {
#pragma unroll
    for (int r = 0; r < 8; ++r) smred[w][r + 8 * h] = red[r];
  }
  __syncthreads();
#pragma unroll
  for (int r = 0; r < 8; ++r) {
    float s = 0.f;
    for (int w2 = 0; w2 < 8; ++w2) s += smred[w2][r + 8 * h];
    red[r] = 1.0f / s;
  }
#pragma unroll
  for (int tt = 0; tt < 8; ++tt)
#pragma unroll
    for (int r = 0; r < 8; ++r)
      bout[(size_t)(r + 8 * h) * HWP + w * 128 + tt * 16 + lc] =
          (__bf16)(acc[tt].f[r] * red[r]);
}

// ------------------------------------------------------------------
// Kernel 4: o_feat[co,n] = sum_m gP[co,m] * beta[n,m]   (K = 1024)
// Both operands bf16 & contiguous per lane -> async global->LDS double buffer.
// Per thread per K-step: 1 async op for A (16B), 2 for B (32B) = 3 ops.
// ------------------------------------------------------------------
__device__ __forceinline__ void ofeat_stage_async(const __bf16* Gb, const __bf16* Bb,
                                                  int mBase, int nBase, int kb, int t,
                                                  __bf16* ldsA, __bf16* ldsB) {
  {  // A: gP row-major [256][1024]; contiguous 8-elem K segment per task
    const int half = t & 1, id = t >> 1, tm = id >> 5, L = id & 31;
    const int m = tm * 16 + (L & 15), h = L >> 4;
    const int k0 = 16 * h + half * 8;
    async_b128(&ldsA[((tm * 32 + L) << 4) + half * 8],
               Gb + (size_t)(mBase + m) * HWP + kb + k0);
  }
  {  // B: beta[n][m] is B^T in memory -> contiguous 16-elem K segment per task
    const int n = t & 127, kh = t >> 7;
    const __bf16* src = Bb + (size_t)(nBase + n) * HWP + kb + kh * 16;
    __bf16* dst = &ldsB[(((n >> 4) * 32 + (n & 15) + (kh << 4)) << 4)];
    async_b128(dst, src);
    async_b128(dst + 8, src + 8);
  }
}

__global__ __launch_bounds__(256) void ofeat_kernel(const __bf16* __restrict__ G,
                                                    const __bf16* __restrict__ Beta,
                                                    __bf16* __restrict__ out) {
  __shared__ __bf16 ldsA[2][4 * 32 * 16];
  __shared__ __bf16 ldsB[2][8 * 32 * 16];
  const int b = blockIdx.z, nBase = blockIdx.x * 128, mBase = blockIdx.y * 64;
  const __bf16* Gb = G + (size_t)b * 256 * HWP;
  const __bf16* Bb = Beta + (size_t)b * HW * HWP;
  const int t = threadIdx.x, L = t & 31, w = t >> 5, wm = w >> 2, wn = w & 3;
  F8 acc[2][2];
#pragma unroll
  for (int i = 0; i < 2; ++i)
#pragma unroll
    for (int j = 0; j < 2; ++j)
#pragma unroll
      for (int r = 0; r < 8; ++r) acc[i][j].f[r] = 0.f;

  // prologue: prefetch K-step 0 into buffer 0
  ofeat_stage_async(Gb, Bb, mBase, nBase, 0, t, ldsA[0], ldsB[0]);
  int buf = 0;
  for (int kb = 0; kb < HWP; kb += 32, buf ^= 1) {
    if (kb + 32 < HWP) {
      // overlap: issue next K-step's copies, then wait until only those remain
      ofeat_stage_async(Gb, Bb, mBase, nBase, kb + 32, t, ldsA[buf ^ 1], ldsB[buf ^ 1]);
      asm volatile("s_wait_asynccnt 0x3" ::: "memory");
    } else {
      asm volatile("s_wait_asynccnt 0x0" ::: "memory");
    }
    __syncthreads();                       // all waves' copies for `buf` landed
    wave_mma(ldsA[buf], ldsB[buf], wm, wn, L, acc);
    __syncthreads();                       // reads done before buf is refilled
  }
  const int h = L >> 4, lc = L & 15;
#pragma unroll
  for (int i = 0; i < 2; ++i)
#pragma unroll
    for (int j = 0; j < 2; ++j) {
      const int mB = mBase + (wm * 2 + i) * 16;
      const int nC = nBase + (wn * 2 + j) * 16 + lc;
      __bf16* o = out + (size_t)b * 256 * HW + (size_t)(mB + 8 * h) * HW + nC;
#pragma unroll
      for (int r = 0; r < 8; ++r) o[(size_t)r * HW] = (__bf16)acc[i][j].f[r];
    }
}

// ------------------------------------------------------------------
// Kernel 5: out[c2,n] = gamma * sum_co w_o[c2,co]*o_feat[co,n] + x[c2,n]
// ------------------------------------------------------------------
__global__ __launch_bounds__(256) void final_kernel(const float* __restrict__ Wo,
                                                    const __bf16* __restrict__ Of,
                                                    const float* __restrict__ X,
                                                    const float* __restrict__ gammap,
                                                    float* __restrict__ out) {
  __shared__ __bf16 ldsA[4 * 32 * 16];
  __shared__ __bf16 ldsB[8 * 32 * 16];
  const int b = blockIdx.z, nBase = blockIdx.x * 128, mBase = blockIdx.y * 64;
  const __bf16* Ob = Of + (size_t)b * 256 * HW;
  const int t = threadIdx.x, L = t & 31, w = t >> 5, wm = w >> 2, wn = w & 3;
  F8 acc[2][2];
#pragma unroll
  for (int i = 0; i < 2; ++i)
#pragma unroll
    for (int j = 0; j < 2; ++j)
#pragma unroll
      for (int r = 0; r < 8; ++r) acc[i][j].f[r] = 0.f;
  for (int kb = 0; kb < 256; kb += 32) {
    stageA_f32(Wo + (size_t)mBase * 256 + kb, 256, t, ldsA);
    stageB_bf16(Ob + (size_t)kb * HW + nBase, HW, t, ldsB);
    __syncthreads();
    wave_mma(ldsA, ldsB, wm, wn, L, acc);
    __syncthreads();
  }
  const float g = *gammap;
  const int h = L >> 4, lc = L & 15;
#pragma unroll
  for (int i = 0; i < 2; ++i)
#pragma unroll
    for (int j = 0; j < 2; ++j) {
      const int mB = mBase + (wm * 2 + i) * 16;
      const int nC = nBase + (wn * 2 + j) * 16 + lc;
      const size_t base = (size_t)b * CIN * HW + (size_t)(mB + 8 * h) * HW + nC;
#pragma unroll
      for (int r = 0; r < 8; ++r)
        out[base + (size_t)r * HW] = g * acc[i][j].f[r] + X[base + (size_t)r * HW];
    }
}

// ------------------------------------------------------------------
extern "C" void kernel_launch(void* const* d_in, const int* in_sizes, int n_in,
                              void* d_out, int out_size, void* d_ws, size_t ws_size,
                              hipStream_t stream) {
  (void)in_sizes; (void)n_in; (void)out_size; (void)ws_size;
  const float* x       = (const float*)d_in[0];
  const float* w_theta = (const float*)d_in[1];
  const float* w_phi   = (const float*)d_in[2];
  const float* w_g     = (const float*)d_in[3];
  const float* w_o     = (const float*)d_in[4];
  const float* gamma   = (const float*)d_in[5];
  float* outp          = (float*)d_out;

  size_t off = 0;
  auto wsalloc = [&](size_t bytes) {
    char* p = (char*)d_ws + off;
    off += (bytes + 255) & ~(size_t)255;
    return p;
  };
  __bf16* thetaT = (__bf16*)wsalloc((size_t)NB * HW * 64 * 2);   //  4 MB [b][n][c]
  __bf16* phiF   = (__bf16*)wsalloc((size_t)NB * 64 * HW * 2);   //  4 MB [b][c][hw]
  __bf16* phiPT  = (__bf16*)wsalloc((size_t)NB * HWP * 64 * 2);  //  1 MB [b][m][c]
  __bf16* gF     = (__bf16*)wsalloc((size_t)NB * 256 * HW * 2);  // 17 MB [b][c][hw]
  __bf16* gP     = (__bf16*)wsalloc((size_t)NB * 256 * HWP * 2); //  4 MB [b][c][m]
  __bf16* betaB  = (__bf16*)wsalloc((size_t)NB * HW * HWP * 2);  // 67 MB [b][n][m]
  __bf16* ofeat  = (__bf16*)wsalloc((size_t)NB * 256 * HW * 2);  // 17 MB [b][c][n]

  const dim3 blk(256);
  // projections (theta stored transposed for the attention A-operand)
  proj_kernel<true ><<<dim3(32, 1, NB), blk, 0, stream>>>(w_theta, x, thetaT, 64, CIN);
  proj_kernel<false><<<dim3(32, 1, NB), blk, 0, stream>>>(w_phi,   x, phiF,   64, CIN);
  proj_kernel<false><<<dim3(32, 4, NB), blk, 0, stream>>>(w_g,     x, gF,    256, CIN);
  // pooling (phi transposed for contiguous B-fragment loads)
  pool_kernel<<<dim3((NB * 64 * HWP) / 256), blk, 0, stream>>>(phiF, phiPT, 64, 1);
  pool_kernel<<<dim3((NB * 256 * HWP) / 256), blk, 0, stream>>>(gF, gP, 256, 0);
  // fused attention scores + softmax -> beta
  scores_softmax_kernel<<<dim3(HW / 16, NB), blk, 0, stream>>>(thetaT, phiPT, betaB);
  // o_feat = gP x beta^T  (async-LDS double-buffered WMMA GEMM)
  ofeat_kernel<<<dim3(32, 4, NB), blk, 0, stream>>>(gP, betaB, ofeat);
  // out = gamma * (w_o x o_feat) + x
  final_kernel<<<dim3(32, 8, NB), blk, 0, stream>>>(w_o, ofeat, x, gamma, outp);
}
